// Attention_7868380086611
// MI455X (gfx1250) — compile-verified, matrix operand order checked by reading, not run on hardware
//
#include <hip/hip_runtime.h>

// ---------------------------------------------------------------------------
// Feature probes (compile-safe: fall back to synchronous staging)
// ---------------------------------------------------------------------------
#ifndef __has_builtin
#define __has_builtin(x) 0
#endif

#if __has_builtin(__builtin_amdgcn_global_load_async_to_lds_b128)
#define HAS_ASYNC 1
#else
#define HAS_ASYNC 0
#endif

#if __has_builtin(__builtin_amdgcn_tensor_load_to_lds) && \
    __has_builtin(__builtin_amdgcn_s_wait_tensorcnt)
#define HAS_TDM 1
#else
#define HAS_TDM 0
#endif

// ---------------------------------------------------------------------------
// Types / helpers
// ---------------------------------------------------------------------------
typedef __attribute__((ext_vector_type(16))) __bf16 bf16x16;
typedef __attribute__((ext_vector_type(8)))  float  f32x8;
typedef __attribute__((ext_vector_type(4)))  unsigned int u32x4;
typedef __attribute__((ext_vector_type(8)))  int i32x8;
typedef __attribute__((ext_vector_type(4)))  int i32x4;

#define AS_GLOBAL __attribute__((address_space(1)))
#define AS_LDS    __attribute__((address_space(3)))

__device__ inline __bf16 f2bf(float f) {
  union { float f; unsigned u; } x; x.f = f;
  unsigned r = (x.u + 0x7FFFu + ((x.u >> 16) & 1u)) >> 16;
  unsigned short s = (unsigned short)r;
  __bf16 b; __builtin_memcpy(&b, &s, 2);
  return b;
}
__device__ inline float bf2f(__bf16 b) {
  unsigned short s; __builtin_memcpy(&s, &b, 2);
  union { unsigned u; float f; } x; x.u = ((unsigned)s) << 16;
  return x.f;
}

__device__ inline f32x8 wmma_bf16(bf16x16 a, bf16x16 b, f32x8 c) {
  return __builtin_amdgcn_wmma_f32_16x16x32_bf16(false, a, false, b, (short)0, c,
                                                 false, false);
}

// 16-byte global -> LDS copy. Async (ASYNCcnt, no VGPR round trip) when the
// toolchain exposes the gfx1250 builtin; else plain load/store.
__device__ inline void async_copy16(const void* gsrc, void* ldst) {
#if HAS_ASYNC
  __builtin_amdgcn_global_load_async_to_lds_b128(
      (AS_GLOBAL i32x4*)gsrc, (AS_LDS i32x4*)ldst, 0, 0);
#else
  *(uint4*)ldst = *(const uint4*)gsrc;
#endif
}
__device__ inline void lds_zero16(void* ldst) {
  uint4 z{0u, 0u, 0u, 0u};
  *(uint4*)ldst = z;
}
__device__ inline void async_wait() {
#if HAS_ASYNC
#if __has_builtin(__builtin_amdgcn_s_wait_asynccnt)
  __builtin_amdgcn_s_wait_asynccnt(0);
#else
  asm volatile("s_wait_asynccnt 0" ::: "memory");
#endif
#endif
}

#if HAS_TDM
// Best-effort 2D Tensor-Data-Mover descriptor (D#) per CDNA5 ISA: group0
// holds lds/global address + type=2; group1 holds dims/strides (data_size
// units, bf16 => code 1) and LDS pad config. Groups 2/3 zero (2D tensor).
// This toolchain's builtin takes 6 operands: (g0, g1, g2, g3, g4, cpol).
__device__ inline void tdm_load_2d(const void* gsrc, void* ldst,
                                   unsigned tensor_d0, unsigned tensor_d1,
                                   unsigned tile_d0, unsigned tile_d1,
                                   unsigned d0_stride,
                                   unsigned pad_interval, unsigned pad_amount,
                                   bool pad_en) {
  unsigned long long ga = (unsigned long long)gsrc;
  unsigned lds = (unsigned)(unsigned long long)((AS_LDS char*)ldst);
  u32x4 g0;
  g0.x = 1u;                                               // count=1 (user D#)
  g0.y = lds;                                              // lds_addr
  g0.z = (unsigned)(ga & 0xFFFFFFFFu);                     // global_addr[31:0]
  g0.w = (unsigned)((ga >> 32) & 0x1FFFFFFu) | (2u << 30); // addr[56:32]|type=2
  i32x8 g1;
  g1[0] = (int)((1u << 16) |                               // data_size = 2B
                ((pad_en ? 1u : 0u) << 20) |
                ((pad_interval & 7u) << 22) |
                ((pad_amount & 127u) << 25));
  g1[1] = (int)((tensor_d0 & 0xFFFFu) << 16);
  g1[2] = (int)(((tensor_d0 >> 16) & 0xFFFFu) | ((tensor_d1 & 0xFFFFu) << 16));
  g1[3] = (int)(((tensor_d1 >> 16) & 0xFFFFu) | ((tile_d0 & 0xFFFFu) << 16));
  g1[4] = (int)(tile_d1 & 0xFFFFu);                        // tile_dim1, dim2=0
  g1[5] = (int)d0_stride;                                  // dim0_stride[31:0]
  g1[6] = 0;
  g1[7] = 0;
  i32x4 z4 = (i32x4)0;
  i32x8 z8 = (i32x8)0;
  __builtin_amdgcn_tensor_load_to_lds(g0, g1, z4, z4, z8, 0);
}
__device__ inline void tdm_wait() {
  __builtin_amdgcn_s_wait_tensorcnt(0);
}
#endif

// 16x32 bf16 fragment load from an LDS tile (row stride multiple of 8 elems).
__device__ inline bf16x16 load_frag(const __bf16* base, int ldw, int lane) {
  int r  = lane & 15;
  int hb = (lane >> 4) & 1;
  const __bf16* p = base + r * ldw + hb * 8;
  union { bf16x16 v; uint4 q[2]; } u;
  u.q[0] = *(const uint4*)(p);
  u.q[1] = *(const uint4*)(p + 16);
  return u.v;
}

// ---------------------------------------------------------------------------
// Problem constants
// ---------------------------------------------------------------------------
#define TDIM   2048
#define CDIM   2048
#define NHH    16
#define HSD    128
#define NLQD   1536
#define NLKVD  512
#define DHRD   64
#define DTOT   (NLKVD + DHRD)   // 576

// ---------------------------------------------------------------------------
// f32 -> bf16 convert
// ---------------------------------------------------------------------------
__global__ void cvt_f32_bf16(const float* __restrict__ in, __bf16* __restrict__ out, int n) {
  int i = blockIdx.x * blockDim.x + threadIdx.x;
  if (i < n) out[i] = f2bf(in[i]);
}

// ---------------------------------------------------------------------------
// Generic tiled bf16 WMMA GEMM with LDS double buffering + async staging.
//   TRANS_B = true : C[M,N] = A[M,K] * B[N,K]^T
//   TRANS_B = false: C[M,N] = A[M,K] * B[K,N]
// 256 threads (8 waves), block tile 128x128, K-step 32, batched via blockIdx.z.
// ---------------------------------------------------------------------------
template<bool TRANS_B, bool OUT_F32>
__global__ __launch_bounds__(256)
void gemm_wmma(const __bf16* __restrict__ A, const __bf16* __restrict__ B,
               void* __restrict__ Cp,
               int M, int N, int K, int lda, int ldb, int ldc,
               long long sA, long long sB, long long sC)
{
  __shared__ __align__(16) __bf16 As[2][128][40];
  __shared__ __align__(16) __bf16 Bs[2][128][40];

  const int batch = blockIdx.z;
  A += (long long)batch * sA;
  B += (long long)batch * sB;

  const int m0 = blockIdx.y * 128;
  const int n0 = blockIdx.x * 128;
  const int tid  = threadIdx.x;
  const int lane = tid & 31;
  const int wave = tid >> 5;
  const int wm = wave & 3;
  const int wn = wave >> 2;

  auto stage = [&](int buf, int k0) {
#pragma unroll
    for (int c = tid; c < 512; c += 256) {
      int r = c >> 2, cc = (c & 3) * 8;
      int gr = m0 + r;
      if (gr < M) async_copy16(&A[(long long)gr * lda + k0 + cc], &As[buf][r][cc]);
      else        lds_zero16(&As[buf][r][cc]);
    }
    if (TRANS_B) {
#pragma unroll
      for (int c = tid; c < 512; c += 256) {
        int r = c >> 2, cc = (c & 3) * 8;
        int gn = n0 + r;
        if (gn < N) async_copy16(&B[(long long)gn * ldb + k0 + cc], &Bs[buf][r][cc]);
        else        lds_zero16(&Bs[buf][r][cc]);
      }
    } else {
      // transpose while staging: read along N (coalesced), write [n][k]
#pragma unroll
      for (int c = tid; c < 4096; c += 256) {
        int n = c & 127, kk = c >> 7;
        int gn = n0 + n;
        __bf16 v = f2bf(0.f);
        if (gn < N) v = B[(long long)(k0 + kk) * ldb + gn];
        Bs[buf][n][kk] = v;
      }
    }
  };

  f32x8 acc[2][4];
#pragma unroll
  for (int mi = 0; mi < 2; ++mi)
#pragma unroll
    for (int ni = 0; ni < 4; ++ni) acc[mi][ni] = (f32x8)0.f;

  stage(0, 0);
  async_wait();
  __syncthreads();

  int buf = 0;
  for (int k0 = 0; k0 < K; k0 += 32) {
    if (k0 + 32 < K) stage(buf ^ 1, k0 + 32);   // prefetch next tile

    bf16x16 af[2], bfrag[4];
#pragma unroll
    for (int mi = 0; mi < 2; ++mi)
      af[mi] = load_frag(&As[buf][wm * 32 + mi * 16][0], 40, lane);
#pragma unroll
    for (int ni = 0; ni < 4; ++ni)
      bfrag[ni] = load_frag(&Bs[buf][wn * 64 + ni * 16][0], 40, lane);
#pragma unroll
    for (int mi = 0; mi < 2; ++mi)
#pragma unroll
      for (int ni = 0; ni < 4; ++ni)
        acc[mi][ni] = wmma_bf16(af[mi], bfrag[ni], acc[mi][ni]);

    async_wait();
    __syncthreads();
    buf ^= 1;
  }

  const int col = lane & 15;
  const int rb  = (lane >> 4) * 8;
#pragma unroll
  for (int mi = 0; mi < 2; ++mi)
#pragma unroll
    for (int ni = 0; ni < 4; ++ni)
#pragma unroll
      for (int e = 0; e < 8; ++e) {
        int gr = m0 + wm * 32 + mi * 16 + rb + e;
        int gc = n0 + wn * 64 + ni * 16 + col;
        if (gr < M && gc < N) {
          long long idx = (long long)batch * sC + (long long)gr * ldc + gc;
          if (OUT_F32) ((float*)Cp)[idx] = acc[mi][ni][e];
          else         ((__bf16*)Cp)[idx] = f2bf(acc[mi][ni][e]);
        }
      }
}

// ---------------------------------------------------------------------------
// bf16 transpose: out[C][R] = in[R][C]
// ---------------------------------------------------------------------------
__global__ void transpose_bf16(const __bf16* __restrict__ in, __bf16* __restrict__ out,
                               int R, int C) {
  __shared__ __bf16 tile[32][33];
  int c0 = blockIdx.x * 32, r0 = blockIdx.y * 32;
  for (int i = threadIdx.y; i < 32; i += 8) {
    int r = r0 + i, c = c0 + threadIdx.x;
    tile[i][threadIdx.x] = (r < R && c < C) ? in[(long long)r * C + c] : f2bf(0.f);
  }
  __syncthreads();
  for (int i = threadIdx.y; i < 32; i += 8) {
    int r = r0 + threadIdx.x, c = c0 + i;
    if (r < R && c < C) out[(long long)c * R + r] = tile[threadIdx.x][i];
  }
}

// ---------------------------------------------------------------------------
// RoPE kernels
// ---------------------------------------------------------------------------
__global__ void rope_k(const __bf16* __restrict__ krp, const float* __restrict__ cosT,
                       const float* __restrict__ sinT, __bf16* __restrict__ k_r) {
  int t = blockIdx.x, d = threadIdx.x;
  float re = bf2f(krp[t * DHRD + 2 * d]);
  float im = bf2f(krp[t * DHRD + 2 * d + 1]);
  float c = cosT[t * 32 + d], s = sinT[t * 32 + d];
  k_r[t * DHRD + 2 * d]     = f2bf(re * c - im * s);
  k_r[t * DHRD + 2 * d + 1] = f2bf(re * s + im * c);
}

__global__ void rope_q(const __bf16* __restrict__ qrp, const float* __restrict__ cosT,
                       const float* __restrict__ sinT, __bf16* __restrict__ q_r) {
  int t = blockIdx.x, p = threadIdx.x;
  int h = p >> 5, d = p & 31;
  float re = bf2f(qrp[t * (NHH * DHRD) + h * DHRD + 2 * d]);
  float im = bf2f(qrp[t * (NHH * DHRD) + h * DHRD + 2 * d + 1]);
  float c = cosT[t * 32 + d], s = sinT[t * 32 + d];
  long long o = ((long long)h * TDIM + t) * DHRD + 2 * d;
  q_r[o]     = f2bf(re * c - im * s);
  q_r[o + 1] = f2bf(re * s + im * c);
}

// ---------------------------------------------------------------------------
// Fused causal flash attention over the latent space, double-buffered K/V
// staging (async to LDS; V^T tile optionally via the Tensor Data Mover).
// Q = [q_eff_h | q_r_h] (T x 576), K = [c_kv | k_r] (T x 576), V = c_kv.
// Block: 4 waves x 16 query rows; 32-key blocks; grid (T/64, NH).
// ---------------------------------------------------------------------------
#define FT   16
#define FW   4
#define FBM  (FT * FW)
#define FBN  32
#define LDQ  584   // 576 + 8 pad
#define LDV  40    // 32 + 8 pad

__global__ __launch_bounds__(128)
void flash_mla(const __bf16* __restrict__ q_eff, const __bf16* __restrict__ q_r,
               const __bf16* __restrict__ c_kv, const __bf16* __restrict__ c_kvT,
               const __bf16* __restrict__ k_r, __bf16* __restrict__ y_lat,
               float scale)
{
  __shared__ __align__(16) __bf16 Qs[FBM][LDQ];
  __shared__ __align__(16) __bf16 Ks[2][FBN][LDQ];
  __shared__ __align__(16) __bf16 Vt[2][NLKVD][LDV];
  __shared__ __align__(16) __bf16 Ps[FW][16][LDV];
  __shared__ float mS[FW][16], lS[FW][16], cS[FW][16];

  const int h   = blockIdx.y;
  const int t0  = blockIdx.x * FBM;
  const int tid = threadIdx.x;
  const int wave = tid >> 5;
  const int lane = tid & 31;

  const __bf16* qe = q_eff + (long long)h * TDIM * NLKVD;
  const __bf16* qr = q_r   + (long long)h * TDIM * DHRD;

  auto stageKV = [&](int buf, int s0) {
    for (int c = tid; c < FBN * 72; c += 128) {
      int k = c / 72, cc = c % 72;
      const __bf16* src = (cc < 64)
          ? &c_kv[(long long)(s0 + k) * NLKVD + cc * 8]
          : &k_r[(long long)(s0 + k) * DHRD + (cc - 64) * 8];
      int dst = (cc < 64) ? cc * 8 : (NLKVD + (cc - 64) * 8);
      async_copy16(src, &Ks[buf][k][dst]);
    }
#if HAS_TDM
    // V^T tile: 512 (latent rows) x 32 (keys) from c_kvT, LDS rows padded by
    // 8 bf16 (pad every 16 DWORDs by 4 DWORDs -> interval code 3, amount 3).
    if (wave == 0)
      tdm_load_2d(&c_kvT[s0], &Vt[buf][0][0],
                  /*tensor_d0=*/TDIM, /*tensor_d1=*/NLKVD,
                  /*tile_d0=*/FBN, /*tile_d1=*/NLKVD,
                  /*d0_stride=*/TDIM,
                  /*pad_interval=*/3, /*pad_amount=*/3, /*pad_en=*/true);
#else
    for (int c = tid; c < NLKVD * 4; c += 128) {
      int n = c >> 2, cc = (c & 3) * 8;
      async_copy16(&c_kvT[(long long)n * TDIM + s0 + cc], &Vt[buf][n][cc]);
    }
#endif
  };
  auto stage_wait = [&]() {
    async_wait();
#if HAS_TDM
    if (wave == 0) tdm_wait();
#endif
    __syncthreads();
  };

  // ---- stage Q (64 x 576)
  for (int c = tid; c < FBM * 72; c += 128) {
    int r = c / 72, cc = c % 72;
    const __bf16* src = (cc < 64)
        ? &qe[(long long)(t0 + r) * NLKVD + cc * 8]
        : &qr[(long long)(t0 + r) * DHRD + (cc - 64) * 8];
    int dst = (cc < 64) ? cc * 8 : (NLKVD + (cc - 64) * 8);
    async_copy16(src, &Qs[r][dst]);
  }
  if (tid < FW * 16) {
    mS[tid >> 4][tid & 15] = -1e30f;
    lS[tid >> 4][tid & 15] = 0.f;
  }

  f32x8 O[32];
#pragma unroll
  for (int nt = 0; nt < 32; ++nt) O[nt] = (f32x8)0.f;

  const int qr0 = t0 + wave * FT;
  const int col = lane & 15;
  const int rb  = (lane >> 4) * 8;

  stageKV(0, 0);
  stage_wait();

  int buf = 0;
  for (int s0 = 0; s0 < t0 + FBM; s0 += FBN) {
    if (s0 + FBN < t0 + FBM) stageKV(buf ^ 1, s0 + FBN);   // prefetch next block

    if (s0 <= qr0 + FT - 1) {
      // ---- S = Q * K^T over 576-wide contraction (2 col tiles)
      f32x8 S0 = (f32x8)0.f, S1 = (f32x8)0.f;
      for (int kk = 0; kk < DTOT; kk += 32) {
        bf16x16 a  = load_frag(&Qs[wave * FT][kk], LDQ, lane);
        bf16x16 b0 = load_frag(&Ks[buf][0][kk],  LDQ, lane);
        bf16x16 b1 = load_frag(&Ks[buf][16][kk], LDQ, lane);
        S0 = wmma_bf16(a, b0, S0);
        S1 = wmma_bf16(a, b1, S1);
      }

      // ---- causal mask + scale; per-row running max
      float sv0[8], sv1[8];
#pragma unroll
      for (int e = 0; e < 8; ++e) {
        int tq = qr0 + rb + e;
        float v0 = S0[e], v1 = S1[e];
        v0 = ((s0 + col)      <= tq) ? v0 * scale : -1e30f;
        v1 = ((s0 + 16 + col) <= tq) ? v1 * scale : -1e30f;
        sv0[e] = v0; sv1[e] = v1;
        float mx = fmaxf(v0, v1);
#pragma unroll
        for (int off = 1; off < 16; off <<= 1) mx = fmaxf(mx, __shfl_xor(mx, off, 32));
        if (col == 0) {
          int r = rb + e;
          float mold = mS[wave][r];
          float mnew = fmaxf(mold, mx);
          mS[wave][r] = mnew;
          cS[wave][r] = __expf(mold - mnew);
        }
      }

      // ---- P = exp(S - m), row sums, stage P (A-operand layout)
#pragma unroll
      for (int e = 0; e < 8; ++e) {
        int r = rb + e;
        float mnew = mS[wave][r];
        float p0 = __expf(sv0[e] - mnew);
        float p1 = __expf(sv1[e] - mnew);
        float rs = p0 + p1;
#pragma unroll
        for (int off = 1; off < 16; off <<= 1) rs += __shfl_xor(rs, off, 32);
        if (col == 0) lS[wave][r] = lS[wave][r] * cS[wave][r] + rs;
        Ps[wave][r][col]      = f2bf(p0);
        Ps[wave][r][col + 16] = f2bf(p1);
      }

      // ---- rescale O, then O += P * V
#pragma unroll
      for (int e = 0; e < 8; ++e) {
        float corr = cS[wave][rb + e];
#pragma unroll
        for (int nt = 0; nt < 32; ++nt) O[nt][e] *= corr;
      }
      bf16x16 pf = load_frag(&Ps[wave][0][0], LDV, lane);
#pragma unroll
      for (int nt = 0; nt < 32; ++nt) {
        bf16x16 vf = load_frag(&Vt[buf][nt * 16][0], LDV, lane);
        O[nt] = wmma_bf16(pf, vf, O[nt]);
      }
    }

    stage_wait();
    buf ^= 1;
  }

  // ---- normalize + write y_lat
  float inv[8];
#pragma unroll
  for (int e = 0; e < 8; ++e) inv[e] = 1.f / lS[wave][rb + e];
#pragma unroll
  for (int nt = 0; nt < 32; ++nt)
#pragma unroll
    for (int e = 0; e < 8; ++e) {
      int row = qr0 + rb + e;
      int c   = nt * 16 + col;
      y_lat[((long long)h * TDIM + row) * NLKVD + c] = f2bf(O[nt][e] * inv[e]);
    }
}

// ---------------------------------------------------------------------------
// Host launcher
// ---------------------------------------------------------------------------
extern "C" void kernel_launch(void* const* d_in, const int* in_sizes, int n_in,
                              void* d_out, int out_size, void* d_ws, size_t ws_size,
                              hipStream_t stream) {
  const float* x     = (const float*)d_in[0];
  const float* fcos  = (const float*)d_in[1];
  const float* fsin  = (const float*)d_in[2];
  const float* W_dq  = (const float*)d_in[3];
  const float* W_uq  = (const float*)d_in[4];
  const float* W_dkv = (const float*)d_in[5];
  const float* W_uk  = (const float*)d_in[6];
  const float* W_uv  = (const float*)d_in[7];
  const float* W_qr  = (const float*)d_in[8];
  const float* W_kr  = (const float*)d_in[9];
  const float* W_o   = (const float*)d_in[10];
  float* out = (float*)d_out;

  char* w = (char*)d_ws;
  auto alloc = [&](size_t elems) {
    char* p = w;
    w += (elems * 2 + 255) & ~(size_t)255;
    return (__bf16*)p;
  };
  __bf16* xh    = alloc((size_t)TDIM * CDIM);
  __bf16* Wdqh  = alloc((size_t)NLQD * CDIM);
  __bf16* Wuqh  = alloc((size_t)CDIM * NLQD);
  __bf16* Wdkvh = alloc((size_t)NLKVD * CDIM);
  __bf16* Wukh  = alloc((size_t)CDIM * NLKVD);
  __bf16* Wuvh  = alloc((size_t)CDIM * NLKVD);
  __bf16* Wqrh  = alloc((size_t)NHH * DHRD * NLQD);
  __bf16* Wkrh  = alloc((size_t)DHRD * CDIM);
  __bf16* Woh   = alloc((size_t)CDIM * CDIM);
  __bf16* c_q   = alloc((size_t)TDIM * NLQD);
  __bf16* c_kv  = alloc((size_t)TDIM * NLKVD);
  __bf16* c_kvT = alloc((size_t)NLKVD * TDIM);
  __bf16* krp   = alloc((size_t)TDIM * DHRD);
  __bf16* k_r   = alloc((size_t)TDIM * DHRD);
  __bf16* qrp   = alloc((size_t)TDIM * NHH * DHRD);
  __bf16* q_r   = alloc((size_t)NHH * TDIM * DHRD);
  __bf16* k_eff = alloc((size_t)NHH * NLQD * NLKVD);
  __bf16* q_eff = alloc((size_t)NHH * TDIM * NLKVD);
  __bf16* WoWuv = alloc((size_t)CDIM * NLKVD);
  __bf16* y_lat = alloc((size_t)NHH * TDIM * NLKVD);

  auto cvt = [&](const float* src, __bf16* dst, int n) {
    cvt_f32_bf16<<<(n + 255) / 256, 256, 0, stream>>>(src, dst, n);
  };
  cvt(x,     xh,    TDIM * CDIM);
  cvt(W_dq,  Wdqh,  NLQD * CDIM);
  cvt(W_uq,  Wuqh,  CDIM * NLQD);
  cvt(W_dkv, Wdkvh, NLKVD * CDIM);
  cvt(W_uk,  Wukh,  CDIM * NLKVD);
  cvt(W_uv,  Wuvh,  CDIM * NLKVD);
  cvt(W_qr,  Wqrh,  NHH * DHRD * NLQD);
  cvt(W_kr,  Wkrh,  DHRD * CDIM);
  cvt(W_o,   Woh,   CDIM * CDIM);

  dim3 blk(256);
  // c_q = x * W_dq^T   (2048 x 1536 x 2048)
  gemm_wmma<true, false><<<dim3(12, 16, 1), blk, 0, stream>>>(
      xh, Wdqh, c_q, TDIM, NLQD, CDIM, CDIM, CDIM, NLQD, 0, 0, 0);
  // c_kv = x * W_dkv^T (2048 x 512 x 2048)
  gemm_wmma<true, false><<<dim3(4, 16, 1), blk, 0, stream>>>(
      xh, Wdkvh, c_kv, TDIM, NLKVD, CDIM, CDIM, CDIM, NLKVD, 0, 0, 0);
  // krp = x * W_kr^T   (2048 x 64 x 2048)
  gemm_wmma<true, false><<<dim3(1, 16, 1), blk, 0, stream>>>(
      xh, Wkrh, krp, TDIM, DHRD, CDIM, CDIM, CDIM, DHRD, 0, 0, 0);
  // qrp = c_q * W_qr^T (2048 x 1024 x 1536)
  gemm_wmma<true, false><<<dim3(8, 16, 1), blk, 0, stream>>>(
      c_q, Wqrh, qrp, TDIM, NHH * DHRD, NLQD, NLQD, NLQD, NHH * DHRD, 0, 0, 0);
  // k_eff[h] = Wuq_h (1536x128, lda=2048, off h*128) * Wuk_h (128x512)
  gemm_wmma<false, false><<<dim3(4, 12, NHH), blk, 0, stream>>>(
      Wuqh, Wukh, k_eff, NLQD, NLKVD, HSD, CDIM, NLKVD, NLKVD,
      HSD, (long long)HSD * NLKVD, (long long)NLQD * NLKVD);
  // q_eff[h] = c_q (2048x1536) * k_eff[h] (1536x512)
  gemm_wmma<false, false><<<dim3(4, 16, NHH), blk, 0, stream>>>(
      c_q, k_eff, q_eff, TDIM, NLKVD, NLQD, NLQD, NLKVD, NLKVD,
      0, (long long)NLQD * NLKVD, (long long)TDIM * NLKVD);
  // WoWuv = W_o (2048x2048) * W_uv (2048x512)
  gemm_wmma<false, false><<<dim3(4, 16, 1), blk, 0, stream>>>(
      Woh, Wuvh, WoWuv, CDIM, NLKVD, CDIM, CDIM, NLKVD, NLKVD, 0, 0, 0);

  rope_k<<<TDIM, 32, 0, stream>>>(krp, fcos, fsin, k_r);
  rope_q<<<TDIM, 512, 0, stream>>>(qrp, fcos, fsin, q_r);
  transpose_bf16<<<dim3(NLKVD / 32, TDIM / 32), dim3(32, 8), 0, stream>>>(
      c_kv, c_kvT, TDIM, NLKVD);

  const float scale = 1.0f / sqrtf((float)(HSD + DHRD));
  flash_mla<<<dim3(TDIM / FBM, NHH), dim3(128), 0, stream>>>(
      q_eff, q_r, c_kv, c_kvT, k_r, y_lat, scale);

  // y[:, h*128:(h+1)*128] = y_lat[h] (2048x512) * WoWuv_h^T (128x512)
  gemm_wmma<true, true><<<dim3(1, 16, NHH), blk, 0, stream>>>(
      y_lat, WoWuv, out, TDIM, HSD, NLKVD, NLKVD, NLKVD, CDIM,
      (long long)TDIM * NLKVD, (long long)HSD * NLKVD, (long long)HSD);
}
